// ARC_14817637171251
// MI455X (gfx1250) — matrix operand order repaired and checked by reading, not compiled, not run on hardware
//
#include <hip/hip_runtime.h>
#include <hip/hip_bf16.h>

#define B_ 8
#define SEQ_ 2048
#define STATE_ 128
#define T_ (SEQ_ + 2*STATE_)
#define DIN_ 1024
#define DK_ 128

typedef __attribute__((ext_vector_type(16))) _Float16 v16h;
typedef __attribute__((ext_vector_type(8)))  _Float16 v8h;
typedef __attribute__((ext_vector_type(2)))  _Float16 v2h;
typedef __attribute__((ext_vector_type(8)))  float    v8f;
typedef __attribute__((ext_vector_type(4)))  int      v4i;

#if defined(__has_builtin)
#if __has_builtin(__builtin_amdgcn_global_load_async_to_lds_b128)
#define HAVE_ASYNC_LDS 1
#endif
#endif

__device__ __forceinline__ int lane_id(){ return (int)(threadIdx.x & 31); }

__device__ __forceinline__ v8f wmma16(v16h a, v16h b, v8f c){
  return __builtin_amdgcn_wmma_f32_16x16x32_f16(false, a, false, b, (short)0, c, false, false);
}

// A fragment 16x32 f16; caller passes pointer to this lane's row at k0.
// lanes 0-15: K = 0..15; lanes 16-31: K = 16..31. 16B vector loads.
__device__ __forceinline__ v16h load_a16(const _Float16* p){
  int l = lane_id();
  int base = (l < 16) ? 0 : 16;
  v8h lo = *(const v8h*)(p + base);
  v8h hi = *(const v8h*)(p + 8 + base);
  v16h a;
#pragma unroll
  for (int i=0;i<8;++i){ a[i] = lo[i]; a[8+i] = hi[i]; }
  return a;
}

// B fragment 32x16: 16 contiguous f16 per lane (lane half picks K half).
__device__ __forceinline__ v16h load16h(const _Float16* p){
  v8h lo = *(const v8h*)p;
  v8h hi = *(const v8h*)(p + 8);
  v16h b;
#pragma unroll
  for (int i=0;i<8;++i){ b[i] = lo[i]; b[8+i] = hi[i]; }
  return b;
}

// C/D 16x16 f32: lane&15 = N, VGPR r -> M = r + 8*(lane>=16)
__device__ __forceinline__ void store_c(float* tile, int ld, int col0, v8f c){
  int l = lane_id();
  int half = l >> 4, n = l & 15;
#pragma unroll
  for (int r=0;r<8;++r) tile[(size_t)(r + half*8)*ld + col0 + n] = c[r];
}

// ---------------------------------------------------------------------------
// Prep: convert + TRANSPOSE 9 weight matrices and cope_emb to f16.
//   WT[m][n][k] = W_m[k][n]   (n = output col 0..127, k = input dim 0..1023)
//   copeT[s][d] = cope[d][s]
// B-fragments then read 16 contiguous f16 per lane.
// ---------------------------------------------------------------------------
struct WPtrs { const float* p[10]; };

__global__ void __launch_bounds__(256) kp_wcvt(WPtrs wp, _Float16* __restrict__ dst)
{
  const size_t WSZ = (size_t)DIN_*DK_;
  size_t i = (size_t)blockIdx.x*256 + threadIdx.x;
  float v;
  if (i < 9*WSZ){
    size_t m = i / WSZ, rem = i % WSZ;
    size_t n = rem / DIN_, k = rem % DIN_;
    v = wp.p[m][k*DK_ + n];
  } else {
    size_t rem = i - 9*WSZ;
    size_t s = rem / DK_, d = rem % DK_;
    v = wp.p[9][d*STATE_ + s];
  }
  dst[i] = (_Float16)v;
}

// ---------------------------------------------------------------------------
// Kernel A: segment LayerNorm + q/k/v projection (WMMA) + RoPE on q/k.
// qf stores RoPE(q)*1/sqrt(DK); kf stores RoPE(k); vT stores v transposed
// [b][d][t]; qm/km store pre-RoPE q/k (middle segment, gating path). All f16.
// ---------------------------------------------------------------------------
__global__ void __launch_bounds__(128) ka_ln_proj_rope(
    const float* __restrict__ x,
    const _Float16* __restrict__ wh,      // 9 x [DK][DIN] f16 transposed weights
    const float* __restrict__ gm,  const float* __restrict__ bm,
    const float* __restrict__ gs,  const float* __restrict__ bs,
    const float* __restrict__ ge,  const float* __restrict__ be,
    const int*   __restrict__ offp,
    _Float16* __restrict__ qf, _Float16* __restrict__ kf, _Float16* __restrict__ vT,
    _Float16* __restrict__ qm, _Float16* __restrict__ km)
{
  __shared__ _Float16 As[16][DIN_];
  __shared__ float red[16][8][2];
  __shared__ float stat[16][2];
  __shared__ float qk[2][16][DK_];

  const size_t WSZ = (size_t)DIN_*DK_;
  int b  = blockIdx.y;
  int t0 = blockIdx.x * 16;
  int seg = (t0 < STATE_) ? 0 : ((t0 < STATE_+SEQ_) ? 1 : 2);
  int wsel = (seg==0) ? 3 : ((seg==1) ? 0 : 6);   // ss -> 3..5, mid -> 0..2, se -> 6..8
  const _Float16 *WQ = wh + (size_t)(wsel+0)*WSZ;
  const _Float16 *WK = wh + (size_t)(wsel+1)*WSZ;
  const _Float16 *WV = wh + (size_t)(wsel+2)*WSZ;
  const float *G  = (seg==0)?gs:((seg==1)?gm:ge);
  const float *Bv = (seg==0)?bs:((seg==1)?bm:be);

  // ---- LayerNorm: 8 threads per row, 128 elems each ----
  int row = threadIdx.x >> 3;
  int sub = threadIdx.x & 7;
  const float* xr = x + ((size_t)b*T_ + t0 + row)*DIN_ + sub*128;
  float s0 = 0.f, s1 = 0.f;
#pragma unroll 4
  for (int i=0;i<128;++i){ float v = xr[i]; s0 += v; s1 += v*v; }
  red[row][sub][0] = s0; red[row][sub][1] = s1;
  __syncthreads();
  if (sub == 0){
    float S = 0.f, SS = 0.f;
#pragma unroll
    for (int j=0;j<8;++j){ S += red[row][j][0]; SS += red[row][j][1]; }
    float mean = S * (1.f/DIN_);
    float var  = SS * (1.f/DIN_) - mean*mean;
    stat[row][0] = mean;
    stat[row][1] = rsqrtf(var + 1e-5f);
  }
  __syncthreads();
  {
    float mean = stat[row][0], rstd = stat[row][1];
#pragma unroll 4
    for (int i=0;i<128;++i){
      int d = sub*128 + i;
      As[row][d] = (_Float16)((xr[i]-mean)*rstd*G[d] + Bv[d]);
    }
  }
  __syncthreads();

  // ---- WMMA: 24 output tiles (q,k,v x 8 col tiles), 4 waves x 6 tiles ----
  int w = threadIdx.x >> 5;
  int l = lane_id(); int half = l >> 4, n = l & 15;
  int kb16 = (l < 16) ? 0 : 16;
  v8f vzero = {};
  v8f acc[6];
#pragma unroll
  for (int i=0;i<6;++i) acc[i] = vzero;
  for (int k0=0;k0<DIN_;k0+=32){
    v16h a = load_a16(&As[l & 15][k0]);
#pragma unroll
    for (int i=0;i<6;++i){
      int idx = w + 4*i;
      int mm = idx >> 3, c = idx & 7;
      const _Float16* W = (mm==0)?WQ:((mm==1)?WK:WV);
      v16h bf = load16h(W + (size_t)(c*16 + n)*DIN_ + k0 + kb16);
      acc[i] = wmma16(a, bf, acc[i]);
    }
  }
#pragma unroll
  for (int i=0;i<6;++i){
    int idx = w + 4*i;
    int mm = idx >> 3, c = idx & 7;
    if (mm == 2){
      // vT[b][d][t], d fixed per lane, t = t0 + half*8 + r : one 16B store
      _Float16* dst = vT + ((size_t)b*DK_ + c*16 + n)*T_ + t0 + half*8;
      v8h pk;
#pragma unroll
      for (int r=0;r<8;++r) pk[r] = (_Float16)acc[i][r];
      *(v8h*)dst = pk;
    } else {
      store_c(&qk[mm][0][0], DK_, c*16, acc[i]);
    }
  }
  __syncthreads();

  // ---- RoPE + packed f16 writes (dim pairs -> b32 stores) ----
  int off = offp[0];
  const float L64   = 0.14391156831212772f;   // ln(10000)/64
  const float SCALE = 0.08838834764831845f;   // 1/sqrt(128), folded into qf
#pragma unroll
  for (int it=0; it<4; ++it){
    int id = (int)threadIdx.x + it*128;       // 512 items: 16 rows x 32 dim-pairs
    int r  = id >> 5;
    int d0 = (id & 31) * 2;
    float p = (float)(t0 + r + off);
    float sn0, cs0, sn1, cs1;
    __sincosf(p * __expf(-(float)d0     * L64), &sn0, &cs0);
    __sincosf(p * __expf(-(float)(d0+1) * L64), &sn1, &cs1);
    size_t ro = ((size_t)b*T_ + t0 + r)*DK_;
    float q10 = qk[0][r][d0],   q20 = qk[0][r][d0+64];
    float q11 = qk[0][r][d0+1], q21 = qk[0][r][d0+65];
    v2h qlo, qhi;
    qlo[0] = (_Float16)((q10*cs0 - q20*sn0)*SCALE);
    qlo[1] = (_Float16)((q11*cs1 - q21*sn1)*SCALE);
    qhi[0] = (_Float16)((q20*cs0 + q10*sn0)*SCALE);
    qhi[1] = (_Float16)((q21*cs1 + q11*sn1)*SCALE);
    *(v2h*)(qf + ro + d0)      = qlo;
    *(v2h*)(qf + ro + d0 + 64) = qhi;
    float k10 = qk[1][r][d0],   k20 = qk[1][r][d0+64];
    float k11 = qk[1][r][d0+1], k21 = qk[1][r][d0+65];
    v2h klo, khi;
    klo[0] = (_Float16)(k10*cs0 - k20*sn0);
    klo[1] = (_Float16)(k11*cs1 - k21*sn1);
    khi[0] = (_Float16)(k20*cs0 + k10*sn0);
    khi[1] = (_Float16)(k21*cs1 + k11*sn1);
    *(v2h*)(kf + ro + d0)      = klo;
    *(v2h*)(kf + ro + d0 + 64) = khi;
    if (seg == 1){
      size_t rm = ((size_t)b*SEQ_ + (t0-STATE_) + r)*DK_;
      v2h a0, a1, b0, b1;
      a0[0]=(_Float16)q10; a0[1]=(_Float16)q11;
      a1[0]=(_Float16)q20; a1[1]=(_Float16)q21;
      b0[0]=(_Float16)k10; b0[1]=(_Float16)k11;
      b1[0]=(_Float16)k20; b1[1]=(_Float16)k21;
      *(v2h*)(qm + rm + d0)      = a0;
      *(v2h*)(qm + rm + d0 + 64) = a1;
      *(v2h*)(km + rm + d0)      = b0;
      *(v2h*)(km + rm + d0 + 64) = b1;
    }
  }
}

// ---------------------------------------------------------------------------
// Kernel B1: gates = sigmoid(q k^T) -> f16 plane; logits_int = q@cope -> f32
// ---------------------------------------------------------------------------
__global__ void __launch_bounds__(256) kb_gates(
    const _Float16* __restrict__ qm, const _Float16* __restrict__ km,
    const _Float16* __restrict__ copeT, float* __restrict__ li,
    _Float16* __restrict__ gates)
{
  __shared__ _Float16 qs[16][DK_];
  int b  = blockIdx.y;
  int t0 = blockIdx.x * 16;
  const _Float16* src = qm + ((size_t)b*SEQ_ + t0)*DK_;   // 4KB contiguous tile
#ifdef HAVE_ASYNC_LDS
  {
    __builtin_amdgcn_global_load_async_to_lds_b128(
        (__attribute__((address_space(1))) v4i*)(const_cast<_Float16*>(src) + (size_t)threadIdx.x*8),
        (__attribute__((address_space(3))) v4i*)(&qs[0][0] + (size_t)threadIdx.x*8),
        0, 0);
#if __has_builtin(__builtin_amdgcn_s_wait_asynccnt)
    __builtin_amdgcn_s_wait_asynccnt(0);
#else
    asm volatile("s_wait_asynccnt 0" ::: "memory");
#endif
  }
#else
  ((uint4*)&qs[0][0])[threadIdx.x] = ((const uint4*)src)[threadIdx.x];
#endif
  __syncthreads();

  int w = threadIdx.x >> 5;
  int l = lane_id(); int half = l >> 4, n = l & 15;
  int kb16 = (l < 16) ? 0 : 16;
  v16h a[4];
#pragma unroll
  for (int kc=0;kc<4;++kc) a[kc] = load_a16(&qs[l & 15][kc*32]);
  v8f vzero = {};

  // logits_int: wave w owns column tile w (copeT[s][d]: contiguous in d)
  {
    v8f ci = vzero;
#pragma unroll
    for (int kc=0;kc<4;++kc){
      v16h bf = load16h(copeT + (size_t)(w*16 + n)*DK_ + kc*32 + kb16);
      ci = wmma16(a[kc], bf, ci);
    }
    store_c(li + ((size_t)b*SEQ_ + t0)*DK_, DK_, w*16, ci);
  }
  // gates over all 128 s-tiles, round-robin across 8 waves
  const _Float16* kmb = km + (size_t)b*SEQ_*DK_;
  for (int j = w; j < SEQ_/16; j += 8){
    int s0 = j*16;
    v8f sc = vzero;
#pragma unroll
    for (int kc=0;kc<4;++kc){
      v16h bf = load16h(kmb + (size_t)(s0 + n)*DK_ + kc*32 + kb16);
      sc = wmma16(a[kc], bf, sc);
    }
#pragma unroll
    for (int r=0;r<8;++r){
      float gv = 1.f / (1.f + __expf(-sc[r]));
      gates[((size_t)b*SEQ_ + t0 + half*8 + r)*SEQ_ + s0 + n] = (_Float16)gv;
    }
  }
}

// ---------------------------------------------------------------------------
// Kernel B2: per-row suffix sum of gates -> interpolated CoPE bias (in place)
// ---------------------------------------------------------------------------
__global__ void __launch_bounds__(256) kb_scan(const float* __restrict__ li,
                                               _Float16* __restrict__ bias)
{
  __shared__ float L[STATE_];
  __shared__ float wsum[8];
  int b = blockIdx.y, t = blockIdx.x;
  int lane = lane_id(), wv = threadIdx.x >> 5;
  if (threadIdx.x < STATE_)
    L[threadIdx.x] = li[((size_t)b*SEQ_ + t)*DK_ + threadIdx.x];
  _Float16* rowp = bias + ((size_t)b*SEQ_ + t)*SEQ_;
  float gv[8];
  int base = (int)threadIdx.x * 8;
  {
    v8h hv = *(const v8h*)(rowp + base);        // one 16B load
#pragma unroll
    for (int i=0;i<8;++i) gv[i] = (float)hv[i];
  }
  float tsum = 0.f;
#pragma unroll
  for (int i=0;i<8;++i) tsum += gv[i];
  float sc = tsum;                       // wave inclusive scan of per-thread sums
#pragma unroll
  for (int d=1; d<32; d<<=1){
    float o = __shfl_up(sc, d, 32);
    if (lane >= d) sc += o;
  }
  if (lane == 31) wsum[wv] = sc;
  __syncthreads();
  float pre = 0.f, tot = 0.f;
#pragma unroll
  for (int i=0;i<8;++i){ float v = wsum[i]; tot += v; if (i < wv) pre += v; }
  float excl = pre + sc - tsum;
  float run = 0.f;
#pragma unroll
  for (int i=0;i<8;++i){
    float pos = tot - (excl + run);      // suffix sum including element i
    run += gv[i];
    pos = fminf(pos, (float)(STATE_-1));
    float pf = floorf(pos);
    int fi = (int)pf;
    int ci = (int)ceilf(pos);
    fi = max(0, min(fi, STATE_-1));
    ci = max(0, min(ci, STATE_-1));
    float wf = pos - pf;
    gv[i] = L[ci]*wf + L[fi]*(1.f - wf);
  }
  {
    v8h ov;
#pragma unroll
    for (int i=0;i<8;++i) ov[i] = (_Float16)gv[i];
    *(v8h*)(rowp + base) = ov;                  // one 16B store
  }
}

// ---------------------------------------------------------------------------
// Kernel C: flash attention, causal + interior CoPE bias. One wave per 16
// query rows; s-steps of 32; all operands f16; P through LDS back into WMMA.
// ---------------------------------------------------------------------------
__global__ void __launch_bounds__(256) kc_attn(
    const _Float16* __restrict__ qf, const _Float16* __restrict__ kf,
    const _Float16* __restrict__ vT, const _Float16* __restrict__ bias,
    float* __restrict__ out)
{
  __shared__ _Float16 pst[8][16][32];
  int b = blockIdx.y;
  int w = threadIdx.x >> 5;
  int l = lane_id();
  int half = l >> 4, n = l & 15;
  int abase = half*8;     // C-row offset
  int bbase = half*16;    // B-frag K offset
  int t0 = blockIdx.x*128 + w*16;

  v16h qa[4];
  {
    const _Float16* qrow = qf + ((size_t)b*T_ + t0 + n)*DK_;  // A row = lane&15
#pragma unroll
    for (int kc=0;kc<4;++kc) qa[kc] = load_a16(qrow + kc*32);
  }

  v8f vzero = {};
  float mr[8], lr[8]; v8f acc[8];
#pragma unroll
  for (int r=0;r<8;++r){ mr[r] = -1e30f; lr[r] = 0.f; }
#pragma unroll
  for (int c=0;c<8;++c) acc[c] = vzero;

  int send = t0 + 16;                     // causal: need s <= t0+15
  for (int s0 = 0; s0 < send; s0 += 32){
    v8f st[2];
#pragma unroll
    for (int j=0;j<2;++j){
      v8f sc = vzero;
#pragma unroll
      for (int kc=0;kc<4;++kc){
        v16h bf = load16h(kf + ((size_t)b*T_ + s0 + j*16 + n)*DK_ + kc*32 + bbase);
        sc = wmma16(qa[kc], bf, sc);
      }
      st[j] = sc;
    }
    // causal mask + interior bias (f16 plane)
#pragma unroll
    for (int j=0;j<2;++j){
      int s = s0 + j*16 + n;
      bool smid = (s >= STATE_) && (s < STATE_+SEQ_);
#pragma unroll
      for (int r=0;r<8;++r){
        int t = t0 + abase + r;
        float add;
        if (s > t) add = -1e30f;
        else if (smid && t >= STATE_ && t < STATE_+SEQ_)
          add = (float)bias[((size_t)b*SEQ_ + (t-STATE_))*SEQ_ + (s-STATE_)];
        else add = 0.f;
        st[j][r] += add;
      }
    }
    // online softmax (row = 16 lanes of one half)
    float al[8];
#pragma unroll
    for (int r=0;r<8;++r){
      float mx = fmaxf(st[0][r], st[1][r]);
#pragma unroll
      for (int msk=1; msk<16; msk<<=1) mx = fmaxf(mx, __shfl_xor(mx, msk, 32));
      float nm = fmaxf(mr[r], mx);
      al[r] = __expf(mr[r] - nm);
      float p0 = __expf(st[0][r] - nm);
      float p1 = __expf(st[1][r] - nm);
      st[0][r] = p0; st[1][r] = p1;
      float rs = p0 + p1;
#pragma unroll
      for (int msk=1; msk<16; msk<<=1) rs += __shfl_xor(rs, msk, 32);
      lr[r] = lr[r]*al[r] + rs;
      mr[r] = nm;
    }
#pragma unroll
    for (int c=0;c<8;++c)
#pragma unroll
      for (int r=0;r<8;++r) acc[c][r] *= al[r];

    // transpose P (C layout) -> A fragment via per-wave LDS tile
#pragma unroll
    for (int j=0;j<2;++j)
#pragma unroll
      for (int r=0;r<8;++r)
        pst[w][abase + r][j*16 + n] = (_Float16)st[j][r];
    asm volatile("s_wait_dscnt 0" ::: "memory");   // same-wave LDS ordering
    v16h pa = load_a16(&pst[w][n][0]);
    // P @ V   (vT[b][d][t]: contiguous in s per lane)
#pragma unroll
    for (int c=0;c<8;++c){
      v16h vb = load16h(vT + ((size_t)b*DK_ + c*16 + n)*T_ + s0 + bbase);
      acc[c] = wmma16(pa, vb, acc[c]);
    }
  }

#pragma unroll
  for (int c=0;c<8;++c){
    float* op = out + ((size_t)b*T_ + t0 + abase)*DK_ + c*16 + n;
#pragma unroll
    for (int r=0;r<8;++r) op[(size_t)r*DK_] = acc[c][r] / lr[r];
  }
}

// ---------------------------------------------------------------------------
extern "C" void kernel_launch(void* const* d_in, const int* in_sizes, int n_in,
                              void* d_out, int out_size, void* d_ws, size_t ws_size,
                              hipStream_t stream)
{
  (void)in_sizes; (void)n_in; (void)out_size; (void)ws_size;
  const float* x    = (const float*)d_in[0];
  const float* gm   = (const float*)d_in[10];
  const float* bm   = (const float*)d_in[11];
  const float* gs   = (const float*)d_in[12];
  const float* bs   = (const float*)d_in[13];
  const float* ge   = (const float*)d_in[14];
  const float* be   = (const float*)d_in[15];
  const int*   off  = (const int*)d_in[17];

  WPtrs wp;
  for (int i=0;i<9;++i) wp.p[i] = (const float*)d_in[1+i];  // Wq..Wv_se
  wp.p[9] = (const float*)d_in[16];                         // cope_emb

  const size_t WSZ = (size_t)DIN_*DK_;
  const size_t CSZ = (size_t)DK_*STATE_;
  const size_t NF  = (size_t)B_*T_*DK_;
  const size_t NM  = (size_t)B_*SEQ_*DK_;

  _Float16* wh  = (_Float16*)d_ws;         // 9*WSZ + CSZ f16 transposed weights
  _Float16* qf  = wh + 9*WSZ + CSZ;
  _Float16* kf  = qf + NF;
  _Float16* vT  = kf + NF;
  _Float16* qm  = vT + NF;
  _Float16* km  = qm + NM;
  float*    li  = (float*)(km + NM);       // NM f32
  _Float16* gb  = (_Float16*)(li + NM);    // B*SEQ*SEQ f16 (gates, then bias)
  _Float16* cpt = wh + 9*WSZ;

  {
    size_t total = 9*WSZ + CSZ;
    kp_wcvt<<<dim3((unsigned)(total/256)), 256, 0, stream>>>(wp, wh);
  }
  dim3 gA(T_/16, B_);
  ka_ln_proj_rope<<<gA, 128, 0, stream>>>(x, wh, gm, bm, gs, bs, ge, be,
                                          off, qf, kf, vT, qm, km);
  dim3 gB(SEQ_/16, B_);
  kb_gates<<<gB, 256, 0, stream>>>(qm, km, cpt, li, gb);
  dim3 gS(SEQ_, B_);
  kb_scan<<<gS, 256, 0, stream>>>(li, gb);
  dim3 gC(T_/128, B_);
  kc_attn<<<gC, 256, 0, stream>>>(qf, kf, vT, gb, (float*)d_out);
}
